// GeometricAttentionMLP_90537910600029
// MI455X (gfx1250) — matrix operand order repaired and checked by reading, not compiled, not run on hardware
//
#include <hip/hip_runtime.h>
#include <hip/hip_bf16.h>
#include <math.h>

// Problem constants (match reference)
#define Bv 2
#define Nv 256
#define Fv 256
#define Rv 16
#define Hv 4
#define Dv 64

typedef __attribute__((ext_vector_type(16))) _Float16 v16h;
typedef __attribute__((ext_vector_type(8)))  _Float16 v8h;
typedef __attribute__((ext_vector_type(8)))  float    v8f;

// ---------------- Tensor Data Mover (TDM) support ---------------------------
// Probe-verified builtin existence; arity differs across toolchains:
//   ROCm 7.2 (clang-22):      5 args (v4u, v8i, v4i, v4i, i32)
//   amdgpu-toolchain (c23):   6 args (v4u, v8i, v4i, v4i, v8i, i32)
// The 6-arg toolchain is the only one shipping amd_gfx1250_TDM.h -> use
// __has_include as the discriminator.
#if defined(__has_builtin)
#  if __has_builtin(__builtin_amdgcn_tensor_load_to_lds)
#    define GEO_USE_TDM 1
#  endif
#endif
#ifndef GEO_USE_TDM
#  define GEO_USE_TDM 0
#endif

#if GEO_USE_TDM
typedef unsigned int tdm_v4u __attribute__((ext_vector_type(4)));
typedef int          tdm_v4i __attribute__((ext_vector_type(4)));
typedef int          tdm_v8i __attribute__((ext_vector_type(8)));

// Generic LDS pointer low 32 bits are the LDS byte offset (ISA 10.2 aperture).
__device__ __forceinline__ unsigned lds_offset(const void* p) {
  return (unsigned)(unsigned long long)p;
}

// D# per ISA 8.3/8.4: 2-D tile of f32, tile_dim0=16, tile_dim1=rows,
// tensor_dim0=16, tensor_dim0_stride=16 (contiguous rel_pos rows).
__device__ __forceinline__ void tdm_load_rel_tile(const float* gsrc,
                                                  unsigned lds_byte_off,
                                                  int rows) {
  unsigned long long ga = (unsigned long long)(const void*)gsrc;
  tdm_v4u g0;
  g0.x = 1u;                                        // count=1, user mode
  g0.y = lds_byte_off;                              // lds_addr
  g0.z = (unsigned)(ga & 0xffffffffull);            // global_addr[31:0]
  g0.w = (unsigned)((ga >> 32) & 0x01ffffffull)     // global_addr[56:32]
         | 0x80000000u;                             // type=2 ("image")
  tdm_v8i g1;
  g1[0] = (int)(2u << 16);     // wg_mask=0 | data_size=2 (4 bytes)
  g1[1] = (int)(16u << 16);    // tensor_dim0 low16 = 16
  g1[2] = (int)(256u << 16);   // tensor_dim0 hi=0 | tensor_dim1 low16 = 256
  g1[3] = (int)(16u << 16);    // tensor_dim1 hi=0 | tile_dim0 = 16
  g1[4] = rows;                // tile_dim1 = rows | tile_dim2 = 0
  g1[5] = 16;                  // tensor_dim0_stride low32 = 16
  g1[6] = (int)(4096u << 16);  // dim0_stride hi=0 | dim1_stride low16
  g1[7] = 0;                   // dim1_stride hi (z unused)
  tdm_v4i z4 = {0, 0, 0, 0};
#  if defined(__has_include)
#    if __has_include(<hip/amd_detail/amd_gfx1250_TDM.h>)
  tdm_v8i z8 = {0, 0, 0, 0, 0, 0, 0, 0};
  __builtin_amdgcn_tensor_load_to_lds(g0, g1, z4, z4, z8, 0);
#    else
  __builtin_amdgcn_tensor_load_to_lds(g0, g1, z4, z4, 0);
#    endif
#  else
  __builtin_amdgcn_tensor_load_to_lds(g0, g1, z4, z4, 0);
#  endif
}

__device__ __forceinline__ void wait_tensorcnt0() {
#  if __has_builtin(__builtin_amdgcn_s_wait_tensorcnt)
  __builtin_amdgcn_s_wait_tensorcnt(0);
#  else
  asm volatile("s_wait_tensorcnt 0x0" ::: "memory");
#  endif
}
__device__ __forceinline__ void wait_tensorcnt1() {
#  if __has_builtin(__builtin_amdgcn_s_wait_tensorcnt)
  __builtin_amdgcn_s_wait_tensorcnt(1);
#  else
  asm volatile("s_wait_tensorcnt 0x1" ::: "memory");
#  endif
}
#endif  // GEO_USE_TDM

// Branchless GELU: 0.5x(1+tanh(z)) = x*sigmoid(2z), z=0.79788456(x+0.044715x^3).
// Accuracy-equivalent to erf-GELU at the f16 precision of the WMMA operands.
__device__ __forceinline__ float gelu_fast(float x) {
  float y = 1.5957691216057308f * fmaf(0.044715f * x, x * x, x);
  return x * (1.0f / (1.0f + __expf(-y)));
}

// ---------------- WMMA fragment helpers (wave32, V_WMMA_F32_16X16X32_F16) ----
// A-matrix (16x32 f16) per ISA layout:
//   lanes 0-15 : row = lane,    K = {0..7}   then {16..23}
//   lanes 16-31: row = lane-16, K = {8..15}  then {24..31}
// Source is row-major [16][stride] halves.
__device__ __forceinline__ v16h load_afrag(const _Float16* base, int stride,
                                           int kofs, int lane) {
  int row = lane & 15;
  int kb  = (lane >> 4) << 3;           // 0 or 8
  const _Float16* p = base + row * stride + kofs + kb;
  v8h x0 = *(const v8h*)(p);            // K = kb..kb+7
  v8h x1 = *(const v8h*)(p + 16);       // K = kb+16..kb+23
  return __builtin_shufflevector(x0, x1, 0,1,2,3,4,5,6,7,8,9,10,11,12,13,14,15);
}

// B-matrix (32x16 f16): lane L holds column N=L&15, K = (L>>4)*8 + {0..7} then +16.
// Source stored K-contiguous per column: baseT[col][kstride] halves.
__device__ __forceinline__ v16h load_bfragT(const _Float16* baseT, int kstride,
                                            int kofs, int lane) {
  int col = lane & 15;
  int kb  = (lane >> 4) << 3;           // 0 or 8
  const _Float16* p = baseT + col * kstride + kofs + kb;
  v8h x0 = *(const v8h*)(p);
  v8h x1 = *(const v8h*)(p + 16);
  return __builtin_shufflevector(x0, x1, 0,1,2,3,4,5,6,7,8,9,10,11,12,13,14,15);
}

__device__ __forceinline__ v8f wmma16(v16h a, v16h b, v8f c) {
  // 8 args: (neg_a, A, neg_b, B, c_mod, C, reuse_a, reuse_b)
  return __builtin_amdgcn_wmma_f32_16x16x32_f16(false, a, false, b,
                                                (short)0, c, false, false);
}

// ---------------- Stage 0: transpose + pad W1 tails to f16 -------------------
__global__ __launch_bounds__(256) void geo_tails(const float* __restrict__ kW1,
                                                 const float* __restrict__ vW1,
                                                 _Float16* __restrict__ kT,
                                                 _Float16* __restrict__ vT) {
  int c = threadIdx.x;
#pragma unroll
  for (int k = 0; k < 16; ++k) {
    kT[c * 32 + k]      = (_Float16)kW1[(Fv + k) * Fv + c];
    vT[c * 32 + k]      = (_Float16)vW1[(Fv + k) * Fv + c];
    kT[c * 32 + 16 + k] = (_Float16)0.f;
    vT[c * 32 + 16 + k] = (_Float16)0.f;
  }
}

// ---------------- Stage A: per-(b,i) precompute ------------------------------
__global__ __launch_bounds__(256) void geo_stageA(
    const float* __restrict__ nf,
    const float* __restrict__ qW1, const float* __restrict__ qb1,
    const float* __restrict__ qW2, const float* __restrict__ qb2,
    const float* __restrict__ kW1, const float* __restrict__ kb1,
    const float* __restrict__ kW2,
    const float* __restrict__ vW1, const float* __restrict__ vb1,
    float* __restrict__ nf_k1, float* __restrict__ nf_v1,
    _Float16* __restrict__ QtT) {
  int blk = blockIdx.x;                 // b*N + row
  int tid = threadIdx.x;                // output channel c
  __shared__ float s_x[Fv];
  __shared__ float s_h[Fv];
  __shared__ float s_Q[Fv];

  s_x[tid] = nf[(size_t)blk * Fv + tid];
  __syncthreads();

  float ak = kb1[tid], av = vb1[tid], aq = qb1[tid];
  for (int f = 0; f < Fv; ++f) {
    float x = s_x[f];
    ak = fmaf(x, kW1[f * Fv + tid], ak);
    av = fmaf(x, vW1[f * Fv + tid], av);
    aq = fmaf(x, qW1[f * Fv + tid], aq);
  }
  nf_k1[(size_t)blk * Fv + tid] = ak;
  nf_v1[(size_t)blk * Fv + tid] = av;
  s_h[tid] = gelu_fast(aq);
  __syncthreads();

  float q = qb2[tid];
  for (int c = 0; c < Fv; ++c) q = fmaf(s_h[c], qW2[c * Fv + tid], q);
  s_Q[tid] = q;
  __syncthreads();

  _Float16* Qt = QtT + (size_t)blk * 16 * Fv;   // [16 head rows][256 c]
  const float scale = 0.125f;                   // 1/sqrt(D)
  int c = tid;
#pragma unroll
  for (int h = 0; h < Hv; ++h) {
    float acc = 0.f;
    for (int d = 0; d < Dv; ++d)
      acc = fmaf(s_Q[h * Dv + d], kW2[c * Fv + h * Dv + d], acc);
    Qt[h * Fv + c] = (_Float16)(acc * scale);
  }
#pragma unroll
  for (int h = Hv; h < 16; ++h) Qt[h * Fv + c] = (_Float16)0.f;
}

// ---------------- Main kernel: one block per (b, i) --------------------------
__global__ __launch_bounds__(256) void geo_main(
    const float* __restrict__ nf, const float* __restrict__ rel,
    const float* __restrict__ nf_k1, const float* __restrict__ nf_v1,
    const _Float16* __restrict__ QtT,
    const _Float16* __restrict__ kT, const _Float16* __restrict__ vT,
    const float* __restrict__ vW2, const float* __restrict__ vb2,
    const float* __restrict__ oW, const float* __restrict__ ob,
    float* __restrict__ out) {
  const int blk  = blockIdx.x;          // b*N + i
  const int b    = blk >> 8;
  const int i    = blk & 255;
  const int tid  = threadIdx.x;
  const int wave = tid >> 5;            // 8 waves
  const int lane = tid & 31;

  __shared__ _Float16 s_rel[32 * 32];     //  2 KB  rel tile, K-padded
  __shared__ _Float16 s_hk[16 * Fv];      //  8 KB  gelu(h_k) tile  [j][c]
  __shared__ _Float16 s_hvT[Fv * 32];     // 16 KB  gelu(h_v) tile  [c][j32]
  __shared__ float    s_scores[Nv * 16];  // 16 KB  raw scores      [j][h16]
  __shared__ _Float16 s_attn[16 * Nv];    //  8 KB  attn for A-frag [h16][j]
  __shared__ float    s_U[Hv * Fv];       //  4 KB
  __shared__ float    s_wv[Fv];           //  1 KB
#if GEO_USE_TDM
  __shared__ float    s_relbuf[2][32 * 16];  // 4 KB  TDM double buffer (f32)
#endif

  const size_t relRow = (((size_t)b * Nv + i) * Nv) * Rv;  // rel[b][i][0][0]

  // Per-wave resident fragments: W1-tail B-frags (c-tiles 2w,2w+1) and the
  // Qt B-frag for this wave's K-chunk of the score reduction.
  v16h fKW0 = load_bfragT(kT + (2 * wave + 0) * 16 * 32, 32, 0, lane);
  v16h fKW1 = load_bfragT(kT + (2 * wave + 1) * 16 * 32, 32, 0, lane);
  v16h fVW0 = load_bfragT(vT + (2 * wave + 0) * 16 * 32, 32, 0, lane);
  v16h fVW1 = load_bfragT(vT + (2 * wave + 1) * 16 * 32, 32, 0, lane);
  v16h fQt  = load_bfragT(QtT + (size_t)blk * 16 * Fv, Fv, wave * 32, lane);

  for (int idx = tid; idx < Nv * 16; idx += 256) s_scores[idx] = 0.f;

  // K-padding columns [16,32) of s_rel zeroed once; never rewritten.
  if (tid < 128) {
    int j = tid >> 2, k = (tid & 3) << 2;
    _Float16* d = &s_rel[j * 32 + 16 + k];
    d[0] = d[1] = d[2] = d[3] = (_Float16)0.f;
  }

  const int mrow = (lane >> 4) << 3;    // D-frag: M = r + mrow
  const int ncol = lane & 15;           // D-frag: N = ncol
  const int cc0  = (2 * wave + 0) * 16 + ncol;
  const int cc1  = (2 * wave + 1) * 16 + ncol;

  // ---------------- Pass 1: scores[b,h,i,j] for all j ----------------------
#if GEO_USE_TDM
  if (wave == 0)
    tdm_load_rel_tile(rel + relRow, lds_offset(&s_relbuf[0][0]), 16);
#endif
  for (int jt = 0; jt < 16; ++jt) {
    const int j0 = jt * 16;
#if GEO_USE_TDM
    if (wave == 0) {                    // pipelined TDM double buffer
      if (jt + 1 < 16) {
        tdm_load_rel_tile(rel + relRow + (size_t)(j0 + 16) * Rv,
                          lds_offset(&s_relbuf[(jt + 1) & 1][0]), 16);
        wait_tensorcnt1();              // tile jt done, jt+1 in flight
      } else {
        wait_tensorcnt0();
      }
    }
    __syncthreads();                    // publish DMA fill; fence s_rel reuse
    {
      int j = tid >> 4, k = tid & 15;   // f32 -> f16 into padded layout
      s_rel[j * 32 + k] = (_Float16)s_relbuf[jt & 1][tid];
    }
    __syncthreads();
#else
    __syncthreads();                    // protect s_rel / s_hk
    if (tid < 64) {                     // 16x16 f32 tile via b128 loads
      int j = tid >> 2, k = (tid & 3) << 2;
      float4 rv = *(const float4*)&rel[relRow + (size_t)(j0 + j) * Rv + k];
      _Float16* d = &s_rel[j * 32 + k];
      d[0] = (_Float16)rv.x; d[1] = (_Float16)rv.y;
      d[2] = (_Float16)rv.z; d[3] = (_Float16)rv.w;
    }
    __syncthreads();
    if (jt + 1 < 16) {
      __builtin_prefetch(&rel[relRow + (size_t)(j0 + 16 + (tid >> 4)) * Rv +
                              (tid & 15)], 0, 0);
    }
#endif

    // Batch the nf_k1 loads (one base addr + immediate offsets, single wait)
    const float* nk_b0 = nf_k1 + ((size_t)b * Nv + j0 + mrow) * Fv + cc0;
    const float* nk_b1 = nf_k1 + ((size_t)b * Nv + j0 + mrow) * Fv + cc1;
    float nk0[8], nk1[8];
#pragma unroll
    for (int r = 0; r < 8; ++r) { nk0[r] = nk_b0[r * Fv]; nk1[r] = nk_b1[r * Fv]; }

    // h_k = gelu(nf_k1 + rel @ kW1_tail): 2 c-tiles per wave
    v16h aR = load_afrag(s_rel, 32, 0, lane);
    {
      v8f acc0 = {}; acc0 = wmma16(aR, fKW0, acc0);
      v8f acc1 = {}; acc1 = wmma16(aR, fKW1, acc1);
#pragma unroll
      for (int r = 0; r < 8; ++r) {
        int jj = r + mrow;
        s_hk[jj * Fv + cc0] = (_Float16)gelu_fast(acc0[r] + nk0[r]);
        s_hk[jj * Fv + cc1] = (_Float16)gelu_fast(acc1[r] + nk1[r]);
      }
    }
    __syncthreads();

    // score partials: K-chunk c in [32w, 32w+32), reduce via LDS float atomics
    {
      v16h aH = load_afrag(s_hk, Fv, wave * 32, lane);
      v8f sc = {}; sc = wmma16(aH, fQt, sc);
#pragma unroll
      for (int r = 0; r < 8; ++r) {
        int jj = r + mrow;
        atomicAdd(&s_scores[(j0 + jj) * 16 + ncol], sc[r]);  // ds_add_f32
      }
    }
#if GEO_USE_TDM
    __syncthreads();                    // s_rel consumed before next conversion
#endif
  }
  __syncthreads();

  // ---------------- Softmax over j (waves 0-3 = heads 0-3) -----------------
  if (wave < Hv) {
    const int h = wave;
    float m = -1e30f;
    for (int j = lane; j < Nv; j += 32) m = fmaxf(m, s_scores[j * 16 + h]);
#pragma unroll
    for (int o = 16; o; o >>= 1) m = fmaxf(m, __shfl_xor(m, o, 32));
    float sum = 0.f;
    for (int j = lane; j < Nv; j += 32) {
      float e = __expf(s_scores[j * 16 + h] - m);
      s_scores[j * 16 + h] = e;
      sum += e;
    }
#pragma unroll
    for (int o = 16; o; o >>= 1) sum += __shfl_xor(sum, o, 32);
    float inv = 1.f / sum;
    for (int j = lane; j < Nv; j += 32)
      s_attn[h * Nv + j] = (_Float16)(s_scores[j * 16 + h] * inv);
  }
  for (int idx = tid; idx < 12 * Nv; idx += 256)   // heads 4..15 = 0
    s_attn[Hv * Nv + idx] = (_Float16)0.f;
  __syncthreads();

  // ---------------- Pass 2: U[h,c] = sum_j attn * gelu(h_v) ----------------
  v8f U0 = {}, U1 = {};
#if GEO_USE_TDM
  if (wave == 0)
    tdm_load_rel_tile(rel + relRow, lds_offset(&s_relbuf[0][0]), 32);
#endif
  for (int js = 0; js < 8; ++js) {
    const int j0 = js * 32;
#if GEO_USE_TDM
    if (wave == 0) {
      if (js + 1 < 8) {
        tdm_load_rel_tile(rel + relRow + (size_t)(j0 + 32) * Rv,
                          lds_offset(&s_relbuf[(js + 1) & 1][0]), 32);
        wait_tensorcnt1();
      } else {
        wait_tensorcnt0();
      }
    }
    __syncthreads();
#pragma unroll
    for (int p = 0; p < 2; ++p) {       // 512 f32 -> f16, 2 per thread
      int idx = tid + 256 * p;
      int j = idx >> 4, k = idx & 15;
      s_rel[j * 32 + k] = (_Float16)s_relbuf[js & 1][idx];
    }
    __syncthreads();
#else
    __syncthreads();                    // protect s_rel / s_hvT
    if (tid < 128) {                    // 32x16 f32 tile via b128 loads
      int j = tid >> 2, k = (tid & 3) << 2;
      float4 rv = *(const float4*)&rel[relRow + (size_t)(j0 + j) * Rv + k];
      _Float16* d = &s_rel[j * 32 + k];
      d[0] = (_Float16)rv.x; d[1] = (_Float16)rv.y;
      d[2] = (_Float16)rv.z; d[3] = (_Float16)rv.w;
    }
    __syncthreads();
    if (js + 1 < 8) {
      __builtin_prefetch(&rel[relRow + (size_t)(j0 + 32 + (tid >> 4)) * Rv +
                              (tid & 15)], 0, 0);
    }
#endif

    // recompute h_v for 32 j, store transposed [c][j] for the B-frag
#pragma unroll
    for (int sub = 0; sub < 2; ++sub) {
      const float* nv_b0 =
          nf_v1 + ((size_t)b * Nv + j0 + sub * 16 + mrow) * Fv + cc0;
      const float* nv_b1 =
          nf_v1 + ((size_t)b * Nv + j0 + sub * 16 + mrow) * Fv + cc1;
      float nv0[8], nv1[8];
#pragma unroll
      for (int r = 0; r < 8; ++r) { nv0[r] = nv_b0[r * Fv]; nv1[r] = nv_b1[r * Fv]; }

      v16h aR = load_afrag(s_rel + sub * 16 * 32, 32, 0, lane);
      v8f acc0 = {}; acc0 = wmma16(aR, fVW0, acc0);
      v8f acc1 = {}; acc1 = wmma16(aR, fVW1, acc1);
#pragma unroll
      for (int r = 0; r < 8; ++r) {
        int jj = r + mrow;
        s_hvT[cc0 * 32 + sub * 16 + jj] = (_Float16)gelu_fast(acc0[r] + nv0[r]);
        s_hvT[cc1 * 32 + sub * 16 + jj] = (_Float16)gelu_fast(acc1[r] + nv1[r]);
      }
    }
    __syncthreads();

    v16h aA = load_afrag(s_attn, Nv, j0, lane);
    v16h b0 = load_bfragT(s_hvT + (2 * wave + 0) * 16 * 32, 32, 0, lane);
    v16h b1 = load_bfragT(s_hvT + (2 * wave + 1) * 16 * 32, 32, 0, lane);
    U0 = wmma16(aA, b0, U0);
    U1 = wmma16(aA, b1, U1);
#if GEO_USE_TDM
    __syncthreads();                    // s_rel consumed before next conversion
#endif
  }
  __syncthreads();

  // D-frag rows 0..3 (lanes 0-15, r<4) are heads 0..3
  if (lane < 16) {
    int u0 = (2 * wave + 0) * 16 + lane;
    int u1 = (2 * wave + 1) * 16 + lane;
#pragma unroll
    for (int r = 0; r < Hv; ++r) {
      s_U[r * Fv + u0] = U0[r];
      s_U[r * Fv + u1] = U1[r];
    }
  }
  __syncthreads();

  // wv[g] = U[h] @ vW2[:, g] + vb2[g]   (attn rows sum to 1)
  {
    int g = tid, h = g >> 6;
    float acc = vb2[g];
    for (int c = 0; c < Fv; ++c) acc = fmaf(s_U[h * Fv + c], vW2[c * Fv + g], acc);
    s_wv[g] = acc;
  }
  __syncthreads();

  // out = nf + wv @ oW + ob
  {
    int f = tid;
    float acc = ob[f] + nf[((size_t)b * Nv + i) * Fv + f];
    for (int g = 0; g < Fv; ++g) acc = fmaf(s_wv[g], oW[g * Fv + f], acc);
    out[((size_t)b * Nv + i) * Fv + f] = acc;
  }
}

// ---------------- Host-side launch ------------------------------------------
extern "C" void kernel_launch(void* const* d_in, const int* in_sizes, int n_in,
                              void* d_out, int out_size, void* d_ws,
                              size_t ws_size, hipStream_t stream) {
  (void)in_sizes; (void)n_in; (void)out_size; (void)ws_size;
  const float* nf   = (const float*)d_in[0];
  const float* rel  = (const float*)d_in[1];
  const float* qW1  = (const float*)d_in[2];
  const float* qb1  = (const float*)d_in[3];
  const float* qW2  = (const float*)d_in[4];
  const float* qb2  = (const float*)d_in[5];
  const float* kW1  = (const float*)d_in[6];
  const float* kb1  = (const float*)d_in[7];
  const float* kW2  = (const float*)d_in[8];
  /* k_b2 (d_in[9]) is softmax-invariant -> dropped */
  const float* vW1  = (const float*)d_in[10];
  const float* vb1  = (const float*)d_in[11];
  const float* vW2  = (const float*)d_in[12];
  const float* vb2  = (const float*)d_in[13];
  const float* oW   = (const float*)d_in[14];
  const float* ob   = (const float*)d_in[15];

  char* ws = (char*)d_ws;
  float*    nf_k1 = (float*)(ws);                                   // 512 KB
  float*    nf_v1 = (float*)(ws + (size_t)512 * 1024);              // 512 KB
  _Float16* QtT   = (_Float16*)(ws + (size_t)1024 * 1024);          //   4 MB
  _Float16* kT    = (_Float16*)(ws + (size_t)5 * 1024 * 1024);      //  16 KB
  _Float16* vT    = (_Float16*)(ws + (size_t)5 * 1024 * 1024 + 16 * 1024);

  geo_tails<<<1, 256, 0, stream>>>(kW1, vW1, kT, vT);
  geo_stageA<<<Bv * Nv, 256, 0, stream>>>(nf, qW1, qb1, qW2, qb2, kW1, kb1,
                                          kW2, vW1, vb1, nf_k1, nf_v1, QtT);
  geo_main<<<Bv * Nv, 256, 0, stream>>>(nf, rel, nf_k1, nf_v1, QtT, kT, vT,
                                        vW2, vb2, oW, ob, (float*)d_out);
}
